// RegionLoss_2D_61263413510181
// MI455X (gfx1250) — matrix operand order repaired and checked by reading, not compiled, not run on hardware
//
#include <hip/hip_runtime.h>
#include <hip/hip_bf16.h>

typedef __attribute__((ext_vector_type(16))) _Float16 v16h;
typedef __attribute__((ext_vector_type(8)))  _Float16 v8h;
typedef __attribute__((ext_vector_type(8)))  float    v8f;

#define NROW   8192           // 8 * 32 * 32 rows
#define CDIM   64             // channels (K dimension)
#define LDSPAD 72             // row stride in halfs: 144B = 36 dwords, conflict-free for b128
#define TI     128            // block tile rows
#define TJ     64             // block tile cols

// ---------------------------------------------------------------------------
// Kernel 1: 4x4 average pool + row L2-normalize + fp16 convert.
// One block per output row n (b*1024 + y*32 + x); thread c = channel.
// ---------------------------------------------------------------------------
__global__ __launch_bounds__(64) void pool_norm_kernel(const float* __restrict__ p,
                                                       _Float16* __restrict__ X) {
    const int n = blockIdx.x;          // 0..8191
    const int c = threadIdx.x;         // 0..63
    const int b  = n >> 10;
    const int yx = n & 1023;
    const int y  = yx >> 5;
    const int x  = yx & 31;

    const float* base = p + ((((size_t)b * 64 + c) * 128 + (size_t)y * 4) * 128 + (size_t)x * 4);
    float s = 0.f;
#pragma unroll
    for (int dy = 0; dy < 4; ++dy) {
#pragma unroll
        for (int dx = 0; dx < 4; ++dx) s += base[dy * 128 + dx];
    }
    const float pooled = s * (1.0f / 16.0f);

    __shared__ float sh[64];
    sh[c] = pooled * pooled;
    __syncthreads();
#pragma unroll
    for (int off = 32; off > 0; off >>= 1) {
        if (c < off) sh[c] += sh[c + off];
        __syncthreads();
    }
    const float norm = sqrtf(sh[0]);
    const float inv  = 1.0f / fmaxf(norm, 1e-8f);
    X[(size_t)n * CDIM + c] = (_Float16)(pooled * inv);
}

// ---------------------------------------------------------------------------
// Fragment loaders per CDNA5 16-bit WMMA VGPR layouts (CDNA5 ISA).
// A (16x32, MxK): lane m=l&15, half=l>>4; halfs 0..7  = K[8*half    .. +8)
//                                          halfs 8..15 = K[16+8*half .. +8)
// B (32x16, KxN): lane n=l&15, half=l>>4; halfs 0..7  = K[8*half    .. +8)
//                                          halfs 8..15 = K[16+8*half .. +8)
// Each run is 16B-aligned contiguous LDS -> ds_load_b128.
// ---------------------------------------------------------------------------
__device__ __forceinline__ v16h load_fragA(const _Float16 (*s)[LDSPAD], int row, int kk, int half) {
    const int k1 = kk * 32 + 8 * half;
    v8h lo = *(const v8h*)&s[row][k1];
    v8h hi = *(const v8h*)&s[row][k1 + 16];
    v16h r;
#pragma unroll
    for (int i = 0; i < 8; ++i) { r[i] = lo[i]; r[i + 8] = hi[i]; }
    return r;
}

__device__ __forceinline__ v16h load_fragB(const _Float16 (*s)[LDSPAD], int row, int kk, int half) {
    const int k1 = kk * 32 + 8 * half;
    v8h lo = *(const v8h*)&s[row][k1];
    v8h hi = *(const v8h*)&s[row][k1 + 16];
    v16h r;
#pragma unroll
    for (int i = 0; i < 8; ++i) { r[i] = lo[i]; r[i + 8] = hi[i]; }
    return r;
}

// ---------------------------------------------------------------------------
// Kernel 2: fused dual-Gram + squared-difference reduction.
// Block (bx,by) owns a 128x64 tile of the 8192x8192 sim matrices.
// 8 wave32s in a 4x2 grid; each wave owns a 32x32 region (2x2 subtiles) with
// full register blocking: A fragments reused across the 2 column tiles, B
// fragments across the 2 row tiles -> 16 WMMA per wave, 2 ds_load_b128/wmma.
// ---------------------------------------------------------------------------
__global__ __launch_bounds__(256) void gram_diff_kernel(const _Float16* __restrict__ Xa,
                                                        const _Float16* __restrict__ Xb,
                                                        float* __restrict__ out) {
    __shared__ __align__(16) _Float16 sAi[TI][LDSPAD];  // Xa rows of row-block
    __shared__ __align__(16) _Float16 sBi[TI][LDSPAD];  // Xb rows of row-block
    __shared__ __align__(16) _Float16 sAj[TJ][LDSPAD];  // Xa rows of col-block
    __shared__ __align__(16) _Float16 sBj[TJ][LDSPAD];  // Xb rows of col-block
    __shared__ float red;

    const int t    = threadIdx.x;
    const int rowI = blockIdx.y * TI;
    const int rowJ = blockIdx.x * TJ;
    if (t == 0) red = 0.f;

    // Cooperative tile staging: rows are 128B = 8 x 16B chunks.
#pragma unroll
    for (int ch = t; ch < TI * 8; ch += 256) {
        const int r = ch >> 3;
        const int o = (ch & 7) * 8;
        *(v8h*)&sAi[r][o] = *(const v8h*)&Xa[(size_t)(rowI + r) * CDIM + o];
        *(v8h*)&sBi[r][o] = *(const v8h*)&Xb[(size_t)(rowI + r) * CDIM + o];
    }
#pragma unroll
    for (int ch = t; ch < TJ * 8; ch += 256) {
        const int r = ch >> 3;
        const int o = (ch & 7) * 8;
        *(v8h*)&sAj[r][o] = *(const v8h*)&Xa[(size_t)(rowJ + r) * CDIM + o];
        *(v8h*)&sBj[r][o] = *(const v8h*)&Xb[(size_t)(rowJ + r) * CDIM + o];
    }
    __syncthreads();

    const int w    = t >> 5;            // wave id (wave32), 0..7 in 4x2 grid
    const int lane = t & 31;
    const int m    = lane & 15;
    const int half = lane >> 4;
    const int ti0  = (w >> 1) * 32;     // wave's 32x32 region
    const int tj0  = (w & 1) * 32;

    v8f accA[2][2] = {};
    v8f accB[2][2] = {};

#pragma unroll
    for (int kk = 0; kk < 2; ++kk) {
        // Hoisted fragments: 8 loads feed 8 WMMAs this kk step.
        v16h aA0 = load_fragA(sAi, ti0 +      m, kk, half);
        v16h aA1 = load_fragA(sAi, ti0 + 16 + m, kk, half);
        v16h bA0 = load_fragB(sAj, tj0 +      m, kk, half);
        v16h bA1 = load_fragB(sAj, tj0 + 16 + m, kk, half);
        v16h aB0 = load_fragA(sBi, ti0 +      m, kk, half);
        v16h aB1 = load_fragA(sBi, ti0 + 16 + m, kk, half);
        v16h bB0 = load_fragB(sBj, tj0 +      m, kk, half);
        v16h bB1 = load_fragB(sBj, tj0 + 16 + m, kk, half);

        accA[0][0] = __builtin_amdgcn_wmma_f32_16x16x32_f16(false, aA0, false, bA0, (short)0, accA[0][0], false, false);
        accA[0][1] = __builtin_amdgcn_wmma_f32_16x16x32_f16(false, aA0, false, bA1, (short)0, accA[0][1], false, false);
        accA[1][0] = __builtin_amdgcn_wmma_f32_16x16x32_f16(false, aA1, false, bA0, (short)0, accA[1][0], false, false);
        accA[1][1] = __builtin_amdgcn_wmma_f32_16x16x32_f16(false, aA1, false, bA1, (short)0, accA[1][1], false, false);
        accB[0][0] = __builtin_amdgcn_wmma_f32_16x16x32_f16(false, aB0, false, bB0, (short)0, accB[0][0], false, false);
        accB[0][1] = __builtin_amdgcn_wmma_f32_16x16x32_f16(false, aB0, false, bB1, (short)0, accB[0][1], false, false);
        accB[1][0] = __builtin_amdgcn_wmma_f32_16x16x32_f16(false, aB1, false, bB0, (short)0, accB[1][0], false, false);
        accB[1][1] = __builtin_amdgcn_wmma_f32_16x16x32_f16(false, aB1, false, bB1, (short)0, accB[1][1], false, false);
    }

    float local = 0.f;
#pragma unroll
    for (int si = 0; si < 2; ++si) {
#pragma unroll
        for (int sj = 0; sj < 2; ++sj) {
#pragma unroll
            for (int i = 0; i < 8; ++i) {
                const float d = accA[si][sj][i] - accB[si][sj][i];
                local += d * d;
            }
        }
    }

    // wave32 reduction, then block reduction, then one global atomic.
#pragma unroll
    for (int off = 16; off > 0; off >>= 1) local += __shfl_down(local, off, 32);
    if (lane == 0) atomicAdd(&red, local);
    __syncthreads();
    if (t == 0) atomicAdd(out, red * (1.0f / (8192.0f * 8192.0f)));
}

__global__ void zero_out_kernel(float* o) { o[0] = 0.0f; }

// ---------------------------------------------------------------------------
extern "C" void kernel_launch(void* const* d_in, const int* in_sizes, int n_in,
                              void* d_out, int out_size, void* d_ws, size_t ws_size,
                              hipStream_t stream) {
    const float* p1 = (const float*)d_in[0];
    const float* p2 = (const float*)d_in[1];
    float* out = (float*)d_out;

    _Float16* Xa = (_Float16*)d_ws;
    _Float16* Xb = (_Float16*)((char*)d_ws + (size_t)NROW * CDIM * sizeof(_Float16));

    pool_norm_kernel<<<NROW, 64, 0, stream>>>(p1, Xa);
    pool_norm_kernel<<<NROW, 64, 0, stream>>>(p2, Xb);
    zero_out_kernel<<<1, 1, 0, stream>>>(out);

    dim3 grid(NROW / TJ, NROW / TI);   // 128 x 64 blocks
    gram_diff_kernel<<<grid, 256, 0, stream>>>(Xa, Xb, out);
}
